// BranchingGNN_57801669869678
// MI455X (gfx1250) — matrix operand order repaired and checked by reading, not compiled, not run on hardware
//
#include <hip/hip_runtime.h>

#define N_ITEMS 50000
#define M_PATS  20000
#define E_EDGES 1000000
#define DIM_IN  64
#define HDIM    128
#define ROUNDS  2

typedef __attribute__((ext_vector_type(16))) __bf16 v16bf;
typedef __attribute__((ext_vector_type(8)))  float  v8f;

union Frag {
    v16bf v;
    unsigned int u[8];
};

// ---------------------------------------------------------------------------
// Zero-fill (workspace is poisoned by the harness; segment_sum needs zeros).
// ---------------------------------------------------------------------------
__global__ void fill_zero_kernel(float* __restrict__ p, long n) {
    long i = (long)blockIdx.x * blockDim.x + threadIdx.x;
    if (i < n) p[i] = 0.0f;
}

// ---------------------------------------------------------------------------
// Weight pre-pass: W f32 [K,H] -> Wt bf16 [H,K] (transposed, so B-fragment
// k-pairs become contiguous 32-bit loads).
// ---------------------------------------------------------------------------
__global__ void wt_convert_kernel(const float* __restrict__ W,
                                  __bf16* __restrict__ Wt, int K, int H) {
    int i = blockIdx.x * blockDim.x + threadIdx.x;
    if (i < K * H) {
        int k = i / H, h = i - k * H;
        Wt[h * K + k] = (__bf16)W[i];
    }
}

// ---------------------------------------------------------------------------
// segment_sum over edges: dst[sidx[e], :] += src[gidx[e], :]
// One thread per (edge, 4-float chunk): E * (H/4) threads, float4 gather,
// 4x global_atomic_add_f32 scatter. Pure HBM/atomic-bound streaming.
// ---------------------------------------------------------------------------
__global__ void scatter_add_kernel(const float* __restrict__ src,
                                   const int* __restrict__ gidx,
                                   const int* __restrict__ sidx,
                                   float* __restrict__ dst, int E) {
    long t = (long)blockIdx.x * blockDim.x + threadIdx.x;
    int e = (int)(t >> 5);            // 32 chunks of 4 floats cover H=128
    if (e >= E) return;
    int c = ((int)t & 31) << 2;
    int g = gidx[e];
    int s = sidx[e];
    const float4 val = *(const float4*)(src + (long)g * HDIM + c);
    float* d = dst + (long)s * HDIM + c;
    atomicAdd(d + 0, val.x);
    atomicAdd(d + 1, val.y);
    atomicAdd(d + 2, val.z);
    atomicAdd(d + 3, val.w);
}

// ---------------------------------------------------------------------------
// Fused WMMA GEMM: out = epilogue( X[R,K] @ W[K,128] + bias )
//   MODE 0: out = relu(acc + bias)
//   MODE 1: out = relu(res + relu(acc + bias))      (message-update fusion)
// Block: 256 threads = 8 wave32; block tile = 16 rows x 128 cols; wave w
// owns cols [16w, 16w+16). A tile staged via LDS in bf16; B read from the
// pre-transposed bf16 weights. Accumulate f32 via v_wmma_f32_16x16x32_bf16.
// ---------------------------------------------------------------------------
template <int MODE, int K>
__global__ __launch_bounds__(256) void gemm_bias_act_kernel(
    const float* __restrict__ X, const __bf16* __restrict__ Wt,
    const float* __restrict__ bias, const float* __restrict__ res,
    float* __restrict__ out) {
    constexpr int STRIDE_A = K + 8;               // even -> 32-bit aligned pairs
    __shared__ __bf16 Alds[16 * STRIDE_A];

    const int tid  = threadIdx.x;
    const int row0 = blockIdx.x * 16;

    // Cooperative load of the 16 x K A-tile, f32 -> bf16.
    constexpr int TOTAL = 16 * K;
#pragma unroll
    for (int i = tid; i < TOTAL; i += 256) {
        int r = i / K;
        int k = i - r * K;
        Alds[r * STRIDE_A + k] = (__bf16)X[(long)(row0 + r) * K + k];
    }
    __syncthreads();

    const int lane  = tid & 31;
    const int wave  = tid >> 5;
    const int nloc  = lane & 15;
    const int ncol  = wave * 16 + nloc;           // global output column
    const int hi    = (lane & 16) ? 1 : 0;        // upper half-wave
    const int abase = hi ? 16 : 0;                // A k-base (contiguous k = 16h + s, matches B)
    const int bbase = hi ? 16 : 0;                // B k-base (contiguous k = 16h + s, matches A)
    const int m     = nloc;                       // A row owned by this lane

    const __bf16* __restrict__ wcol = Wt + (long)ncol * K;
    const __bf16* __restrict__ arow = &Alds[m * STRIDE_A];

    v8f acc = {};
#pragma unroll
    for (int ks = 0; ks < K; ks += 32) {
        Frag a, b;
#pragma unroll
        for (int v = 0; v < 4; ++v) {
            a.u[v]     = *(const unsigned int*)(arow + ks + abase + 2 * v);
            a.u[v + 4] = *(const unsigned int*)(arow + ks + abase + 8 + 2 * v);
        }
#pragma unroll
        for (int v = 0; v < 8; ++v) {
            b.u[v] = *(const unsigned int*)(wcol + ks + bbase + 2 * v);
        }
        acc = __builtin_amdgcn_wmma_f32_16x16x32_bf16(
            /*neg_a=*/false, a.v, /*neg_b=*/false, b.v,
            /*c_mod=*/(short)0, acc, /*reuse_a=*/false, /*reuse_b=*/false);
    }

    const float bcol = bias[ncol];
#pragma unroll
    for (int v = 0; v < 8; ++v) {
        const int  mr  = v + (hi << 3);           // C/D layout: M = v (+8 upper)
        const long off = (long)(row0 + mr) * HDIM + ncol;
        float val = fmaxf(acc[v] + bcol, 0.0f);
        if (MODE == 1) val = fmaxf(res[off] + val, 0.0f);
        out[off] = val;
    }
}

// ---------------------------------------------------------------------------
// Host-side orchestration (graph-capture safe: kernels only, all on stream).
// ---------------------------------------------------------------------------
extern "C" void kernel_launch(void* const* d_in, const int* in_sizes, int n_in,
                              void* d_out, int out_size, void* d_ws, size_t ws_size,
                              hipStream_t stream) {
    (void)in_sizes; (void)n_in; (void)out_size; (void)ws_size;

    const float* item_feat    = (const float*)d_in[0];
    const float* pattern_feat = (const float*)d_in[1];
    const int*   i_idx        = (const int*)d_in[2];
    const int*   p_idx        = (const int*)d_in[3];
    const float* W_item       = (const float*)d_in[4];
    const float* b_item       = (const float*)d_in[5];
    const float* W_pat        = (const float*)d_in[6];
    const float* b_pat        = (const float*)d_in[7];
    const float* W_i2p        = (const float*)d_in[8];
    const float* b_i2p        = (const float*)d_in[9];
    const float* W_p2i        = (const float*)d_in[10];
    const float* b_p2i        = (const float*)d_in[11];

    float* h_item = (float*)d_out;                        // [N, H]
    float* h_pat  = (float*)d_out + (long)N_ITEMS * HDIM; // [M, H]

    char* ws = (char*)d_ws;
    float* p_msg = (float*)ws;  ws += (size_t)M_PATS  * HDIM * sizeof(float);
    float* i_msg = (float*)ws;  ws += (size_t)N_ITEMS * HDIM * sizeof(float);
    __bf16* Wt_item = (__bf16*)ws;  ws += (size_t)DIM_IN * HDIM * sizeof(__bf16);
    __bf16* Wt_pat  = (__bf16*)ws;  ws += (size_t)DIM_IN * HDIM * sizeof(__bf16);
    __bf16* Wt_i2p  = (__bf16*)ws;  ws += (size_t)HDIM   * HDIM * sizeof(__bf16);
    __bf16* Wt_p2i  = (__bf16*)ws;  ws += (size_t)HDIM   * HDIM * sizeof(__bf16);

    // Weight pre-conversion (f32 [K,H] -> bf16 [H,K]); tiny, runs once per call.
    wt_convert_kernel<<<(DIM_IN * HDIM + 255) / 256, 256, 0, stream>>>(W_item, Wt_item, DIM_IN, HDIM);
    wt_convert_kernel<<<(DIM_IN * HDIM + 255) / 256, 256, 0, stream>>>(W_pat,  Wt_pat,  DIM_IN, HDIM);
    wt_convert_kernel<<<(HDIM  * HDIM + 255) / 256, 256, 0, stream>>>(W_i2p,  Wt_i2p,  HDIM,  HDIM);
    wt_convert_kernel<<<(HDIM  * HDIM + 255) / 256, 256, 0, stream>>>(W_p2i,  Wt_p2i,  HDIM,  HDIM);

    // Input projections: h = relu(X @ W + b).
    gemm_bias_act_kernel<0, DIM_IN><<<N_ITEMS / 16, 256, 0, stream>>>(
        item_feat, Wt_item, b_item, nullptr, h_item);
    gemm_bias_act_kernel<0, DIM_IN><<<M_PATS / 16, 256, 0, stream>>>(
        pattern_feat, Wt_pat, b_pat, nullptr, h_pat);

    const int scatter_blocks = (int)(((long)E_EDGES * 32 + 255) / 256);
    const long pm_elems = (long)M_PATS  * HDIM;
    const long im_elems = (long)N_ITEMS * HDIM;

    for (int r = 0; r < ROUNDS; ++r) {
        // item -> pattern
        fill_zero_kernel<<<(int)((pm_elems + 255) / 256), 256, 0, stream>>>(p_msg, pm_elems);
        scatter_add_kernel<<<scatter_blocks, 256, 0, stream>>>(h_item, i_idx, p_idx, p_msg, E_EDGES);
        // h_pat = relu(h_pat + relu(p_msg @ W_i2p + b_i2p)); fused, in-place.
        gemm_bias_act_kernel<1, HDIM><<<M_PATS / 16, 256, 0, stream>>>(
            p_msg, Wt_i2p, b_i2p, h_pat, h_pat);

        // pattern -> item
        fill_zero_kernel<<<(int)((im_elems + 255) / 256), 256, 0, stream>>>(i_msg, im_elems);
        scatter_add_kernel<<<scatter_blocks, 256, 0, stream>>>(h_pat, p_idx, i_idx, i_msg, E_EDGES);
        // h_item = relu(h_item + relu(i_msg @ W_p2i + b_p2i)); fused, in-place.
        gemm_bias_act_kernel<1, HDIM><<<N_ITEMS / 16, 256, 0, stream>>>(
            i_msg, Wt_p2i, b_p2i, h_item, h_item);
    }
}
